// VQVAE_59150289600649
// MI455X (gfx1250) — matrix operand order repaired and checked by reading, not compiled, not run on hardware
//
#include <hip/hip_runtime.h>
#include <hip/hip_bf16.h>

// ---------------------------------------------------------------------------
// VQ-VAE forward for MI455X (gfx1250, wave32, WMMA).
// bf16 operands pre-converted once; f32 WMMA accumulate; 128x128 block tiles;
// double-buffered LDS fed by gfx1250 async global->LDS copies (ASYNCcnt)
// when available, else register-pipelined b128 staging.
// ---------------------------------------------------------------------------

#define VB   16384
#define VDIN 784
#define VH   400
#define VD   512
#define VK   8192

#define KP1  800     // D_IN=784 -> 800   (K padded to 32)
#define KP2  416     // H=400   -> 416
#define NP_H 512     // N=400/512 -> 512  (N padded to 128)
#define NP_O 896     // N=784 -> 896

typedef __attribute__((ext_vector_type(16))) __bf16 v16bf;
typedef __attribute__((ext_vector_type(8)))  float  v8f;

union FragCast { uint4 u[2]; v16bf v; };

#if __has_builtin(__builtin_amdgcn_global_load_async_to_lds_b128) && \
    __has_builtin(__builtin_amdgcn_s_wait_asynccnt)
#define HAVE_ASYNC 1
#else
#define HAVE_ASYNC 0
#endif

#if HAVE_ASYNC
// builtin signature (from hipcc diagnostic): 4 args,
//   (v4i addrspace(1)*, v4i addrspace(3)*, imm int, imm int)
typedef int b128v __attribute__((vector_size(16)));
typedef __attribute__((address_space(1))) b128v* gp_b128;
typedef __attribute__((address_space(3))) b128v* lp_b128;

static __device__ __forceinline__ void async_b128(const void* g, void* l) {
    __builtin_amdgcn_global_load_async_to_lds_b128((gp_b128)g, (lp_b128)l, 0, 0);
}
#endif

static __device__ __forceinline__ unsigned short f2bf(float f) {
    unsigned u = __float_as_uint(f);
    unsigned r = u + 0x7FFFu + ((u >> 16) & 1u);   // round-to-nearest-even
    return (unsigned short)(r >> 16);
}
static __device__ __forceinline__ unsigned pack_bf2(float a, float b) {
    return (unsigned)f2bf(a) | ((unsigned)f2bf(b) << 16);
}
static __device__ __forceinline__ unsigned long long shfl_xor_u64(unsigned long long v, int mask) {
    unsigned lo = (unsigned)v, hi = (unsigned)(v >> 32);
    lo = __shfl_xor(lo, mask, 32);
    hi = __shfl_xor(hi, mask, 32);
    return ((unsigned long long)hi << 32) | (unsigned long long)lo;
}

#define ACT_RELU   0
#define ACT_NONE   1
#define ACT_SIGBCE 2

// ---------------------------------------------------------------------------
// Pre-conversion kernels (run once per call; tiny vs. the GEMM work)
// ---------------------------------------------------------------------------
__global__ void convert_pad_kernel(const float* __restrict__ src,
                                   unsigned short* __restrict__ dst,
                                   int C, int Cpad, int total4)
{
    int id = blockIdx.x * 256 + threadIdx.x;
    if (id >= total4) return;
    const int cq = Cpad >> 2;
    int r = id / cq, c = (id % cq) * 4;
    const float* s = src + (size_t)r * C;
    float f0 = (c + 0 < C) ? s[c + 0] : 0.f;
    float f1 = (c + 1 < C) ? s[c + 1] : 0.f;
    float f2 = (c + 2 < C) ? s[c + 2] : 0.f;
    float f3 = (c + 3 < C) ? s[c + 3] : 0.f;
    uint2 u; u.x = pack_bf2(f0, f1); u.y = pack_bf2(f2, f3);
    *(uint2*)&dst[(size_t)r * Cpad + c] = u;
}

// W (KxN f32, row-major) -> Wt (Npad x Kpad bf16, [n][k]), zero padded
__global__ void transpose_convert_kernel(const float* __restrict__ W,
                                         unsigned short* __restrict__ Wt,
                                         int K, int N, int Kpad, int total2)
{
    int id = blockIdx.x * 256 + threadIdx.x;
    if (id >= total2) return;
    const int kq = Kpad >> 1;
    int n = id / kq, k = (id % kq) * 2;
    float v0 = (k + 0 < K && n < N) ? W[(size_t)(k + 0) * N + n] : 0.f;
    float v1 = (k + 1 < K && n < N) ? W[(size_t)(k + 1) * N + n] : 0.f;
    *(unsigned*)&Wt[(size_t)n * Kpad + k] = pack_bf2(v0, v1);
}

// ---------------------------------------------------------------------------
// Fused bf16 GEMM: C = act(A @ W + bias).
// Block tile 128x128, K-chunk 32, 8 waves, wave tile 32x64 (2x4 fragments).
// ---------------------------------------------------------------------------
template<int ACT, bool GATHER, bool OUT_F32, bool OUT_B16>
__global__ __launch_bounds__(256)
void gemm_bf16_kernel(const unsigned short* __restrict__ Ab, int strideA,
                      const unsigned long long* __restrict__ gkeys,
                      const unsigned short* __restrict__ Bt, int strideB,
                      const float* __restrict__ bias,
                      float* __restrict__ Cf,
                      unsigned short* __restrict__ Cb, int strideCb,
                      int N, int nk,
                      const float* __restrict__ X,
                      float* __restrict__ partials)
{
    __shared__ __align__(16) unsigned short As[2][128 * 32];
    __shared__ __align__(16) unsigned short Bs[2][128 * 32];
    __shared__ float red[256];

    const int tid  = threadIdx.x;
    const int lane = tid & 31;
    const int wid  = tid >> 5;
    const int wm   = wid >> 1;          // 0..3 (rows of waves)
    const int wn   = wid & 1;           // 0..1 (cols of waves, x64)
    const int m0   = blockIdx.y * 128;
    const int n0   = blockIdx.x * 128;

    const int lrow = tid >> 1;          // staged row, 0..127
    const int half = tid & 1;           // 32B half of the 64B chunk-row

    const unsigned short* aptr;
    if (GATHER) {
        unsigned idx = (unsigned)(gkeys[m0 + lrow] & 0xffffffffull);
        aptr = Ab + (size_t)idx * (size_t)strideA;
    } else {
        aptr = Ab + (size_t)(m0 + lrow) * (size_t)strideA;
    }
    const unsigned short* bptr = Bt + (size_t)(n0 + lrow) * (size_t)strideB;

    v8f acc[2][4] = {};

    const int fr  = lane & 15;
    const int kb  = (lane >> 4) * 8;
    const int kb2 = (lane >> 4) * 16;
    const int eoff = half * 16;                 // element offset of this 32B half
    float bce = 0.f;

#if HAVE_ASYNC
    // ---- async global->LDS staging (ASYNCcnt), double-buffered ----
    {
        const unsigned short* ga = aptr + eoff;
        const unsigned short* gb = bptr + eoff;
        unsigned short* la = &As[0][lrow * 32 + eoff];
        unsigned short* lb = &Bs[0][lrow * 32 + eoff];
        async_b128(ga, la); async_b128(ga + 8, la + 8);
        async_b128(gb, lb); async_b128(gb + 8, lb + 8);
    }
    for (int kc = 0; kc < nk; ++kc) {
        const int buf = kc & 1;
        if (kc + 1 < nk) {
            const int k0 = (kc + 1) << 5;
            const unsigned short* ga = aptr + k0 + eoff;
            const unsigned short* gb = bptr + k0 + eoff;
            unsigned short* la = &As[buf ^ 1][lrow * 32 + eoff];
            unsigned short* lb = &Bs[buf ^ 1][lrow * 32 + eoff];
            async_b128(ga, la); async_b128(ga + 8, la + 8);
            async_b128(gb, lb); async_b128(gb + 8, lb + 8);
            if (kc + 2 < nk) {
                __builtin_prefetch(aptr + ((kc + 2) << 5), 0, 0);
                __builtin_prefetch(bptr + ((kc + 2) << 5), 0, 0);
            }
            __builtin_amdgcn_s_wait_asynccnt(4);   // prev chunk landed (in-order)
        } else {
            __builtin_amdgcn_s_wait_asynccnt(0);
        }
        __syncthreads();

        v16bf af[2], bfq[4];
#pragma unroll
        for (int ms = 0; ms < 2; ++ms) {
            const unsigned short* p = &As[buf][(wm * 32 + ms * 16 + fr) * 32];
            FragCast fc;
            fc.u[0] = *(const uint4*)(p + kb);
            fc.u[1] = *(const uint4*)(p + 16 + kb);
            af[ms] = fc.v;
        }
#pragma unroll
        for (int ns = 0; ns < 4; ++ns) {
            const unsigned short* p = &Bs[buf][(wn * 64 + ns * 16 + fr) * 32];
            FragCast fc;
            fc.u[0] = *(const uint4*)(p + kb);
            fc.u[1] = *(const uint4*)(p + 16 + kb);
            bfq[ns] = fc.v;
        }
#pragma unroll
        for (int ms = 0; ms < 2; ++ms)
#pragma unroll
            for (int ns = 0; ns < 4; ++ns)
                acc[ms][ns] = __builtin_amdgcn_wmma_f32_16x16x32_bf16(
                    false, af[ms], false, bfq[ns],
                    (short)0, acc[ms][ns], false, false);
        __syncthreads();   // buffer we overwrite next iter is now free
    }
#else
    // ---- fallback: register-pipelined staging ----
    uint4 ra0, ra1, rb0, rb1;
    {
        const uint4* ga = (const uint4*)aptr + half * 2;
        ra0 = ga[0]; ra1 = ga[1];
        const uint4* gb = (const uint4*)bptr + half * 2;
        rb0 = gb[0]; rb1 = gb[1];
        uint4* sa = (uint4*)&As[0][lrow * 32 + eoff]; sa[0] = ra0; sa[1] = ra1;
        uint4* sb = (uint4*)&Bs[0][lrow * 32 + eoff]; sb[0] = rb0; sb[1] = rb1;
    }
    __syncthreads();
    for (int kc = 0; kc < nk; ++kc) {
        const int buf = kc & 1;
        if (kc + 1 < nk) {
            const int k0 = (kc + 1) << 5;
            const uint4* ga = (const uint4*)(aptr + k0) + half * 2;
            ra0 = ga[0]; ra1 = ga[1];
            const uint4* gb = (const uint4*)(bptr + k0) + half * 2;
            rb0 = gb[0]; rb1 = gb[1];
            if (kc + 2 < nk) {
                __builtin_prefetch(aptr + ((kc + 2) << 5), 0, 0);
                __builtin_prefetch(bptr + ((kc + 2) << 5), 0, 0);
            }
        }
        v16bf af[2], bfq[4];
#pragma unroll
        for (int ms = 0; ms < 2; ++ms) {
            const unsigned short* p = &As[buf][(wm * 32 + ms * 16 + fr) * 32];
            FragCast fc;
            fc.u[0] = *(const uint4*)(p + kb);
            fc.u[1] = *(const uint4*)(p + 16 + kb);
            af[ms] = fc.v;
        }
#pragma unroll
        for (int ns = 0; ns < 4; ++ns) {
            const unsigned short* p = &Bs[buf][(wn * 64 + ns * 16 + fr) * 32];
            FragCast fc;
            fc.u[0] = *(const uint4*)(p + kb);
            fc.u[1] = *(const uint4*)(p + 16 + kb);
            bfq[ns] = fc.v;
        }
#pragma unroll
        for (int ms = 0; ms < 2; ++ms)
#pragma unroll
            for (int ns = 0; ns < 4; ++ns)
                acc[ms][ns] = __builtin_amdgcn_wmma_f32_16x16x32_bf16(
                    false, af[ms], false, bfq[ns],
                    (short)0, acc[ms][ns], false, false);
        if (kc + 1 < nk) {
            uint4* sa = (uint4*)&As[buf ^ 1][lrow * 32 + eoff]; sa[0] = ra0; sa[1] = ra1;
            uint4* sb = (uint4*)&Bs[buf ^ 1][lrow * 32 + eoff]; sb[0] = rb0; sb[1] = rb1;
        }
        __syncthreads();
    }
#endif

    // ---- epilogue: bias + activation, f32 and/or bf16 outputs ----
#pragma unroll
    for (int ms = 0; ms < 2; ++ms)
#pragma unroll
        for (int ns = 0; ns < 4; ++ns) {
            const int col   = n0 + wn * 64 + ns * 16 + fr;
            const int rbase = m0 + wm * 32 + ms * 16 + (lane >> 4) * 8;
            if (col < N) {
                const float bv = bias[col];
#pragma unroll
                for (int i = 0; i < 8; ++i) {
                    const int row = rbase + i;
                    float v = acc[ms][ns][i] + bv;
                    if (ACT == ACT_RELU) {
                        v = fmaxf(v, 0.f);
                    } else if (ACT == ACT_SIGBCE) {
                        v = 1.f / (1.f + expf(-v));
                        const float pc = fminf(fmaxf(v, 1e-7f), 1.f - 1e-7f);
                        const float xv = X[(size_t)row * (size_t)N + col];
                        bce += xv * logf(pc) + (1.f - xv) * log1pf(-pc);
                    }
                    if (OUT_F32) Cf[(size_t)row * (size_t)N + col] = v;
                    if (OUT_B16) Cb[(size_t)row * (size_t)strideCb + col] = f2bf(v);
                }
            } else if (OUT_B16 && col < strideCb) {
#pragma unroll
                for (int i = 0; i < 8; ++i)
                    Cb[(size_t)(rbase + i) * (size_t)strideCb + col] = 0;
            }
        }

    if (ACT == ACT_SIGBCE) {
        red[tid] = bce;
        __syncthreads();
        for (int s = 128; s > 0; s >>= 1) {
            if (tid < s) red[tid] += red[tid + s];
            __syncthreads();
        }
        if (tid == 0) partials[blockIdx.y * gridDim.x + blockIdx.x] = red[0];
    }
}

// ---------------------------------------------------------------------------
// Distance + argmin: per row b minimize enorm[k] - 2*<z_e[b], emb[k]>.
// ---------------------------------------------------------------------------
__global__ __launch_bounds__(256)
void vq_dist_argmin_kernel(const unsigned short* __restrict__ zeb,
                           const unsigned short* __restrict__ embb,
                           const float* __restrict__ enorm,
                           unsigned long long* __restrict__ keys)
{
    __shared__ __align__(16) unsigned short As[2][128 * 32];
    __shared__ __align__(16) unsigned short Bs[2][128 * 32];

    const int tid  = threadIdx.x;
    const int lane = tid & 31;
    const int wid  = tid >> 5;
    const int wm   = wid >> 1;
    const int wn   = wid & 1;
    const int m0   = blockIdx.y * 128;
    const int n0   = blockIdx.x * 128;
    const int lrow = tid >> 1;
    const int half = tid & 1;

    const unsigned short* aptr = zeb  + (size_t)(m0 + lrow) * VD;
    const unsigned short* bptr = embb + (size_t)(n0 + lrow) * VD;

    v8f acc[2][4] = {};
    const int fr  = lane & 15;
    const int kb  = (lane >> 4) * 8;
    const int kb2 = (lane >> 4) * 16;
    const int eoff = half * 16;
    const int nk  = VD / 32;            // 16

#if HAVE_ASYNC
    {
        const unsigned short* ga = aptr + eoff;
        const unsigned short* gb = bptr + eoff;
        unsigned short* la = &As[0][lrow * 32 + eoff];
        unsigned short* lb = &Bs[0][lrow * 32 + eoff];
        async_b128(ga, la); async_b128(ga + 8, la + 8);
        async_b128(gb, lb); async_b128(gb + 8, lb + 8);
    }
    for (int kc = 0; kc < nk; ++kc) {
        const int buf = kc & 1;
        if (kc + 1 < nk) {
            const int k0 = (kc + 1) << 5;
            const unsigned short* ga = aptr + k0 + eoff;
            const unsigned short* gb = bptr + k0 + eoff;
            unsigned short* la = &As[buf ^ 1][lrow * 32 + eoff];
            unsigned short* lb = &Bs[buf ^ 1][lrow * 32 + eoff];
            async_b128(ga, la); async_b128(ga + 8, la + 8);
            async_b128(gb, lb); async_b128(gb + 8, lb + 8);
            if (kc + 2 < nk) {
                __builtin_prefetch(aptr + ((kc + 2) << 5), 0, 0);
                __builtin_prefetch(bptr + ((kc + 2) << 5), 0, 0);
            }
            __builtin_amdgcn_s_wait_asynccnt(4);
        } else {
            __builtin_amdgcn_s_wait_asynccnt(0);
        }
        __syncthreads();

        v16bf af[2], bfq[4];
#pragma unroll
        for (int ms = 0; ms < 2; ++ms) {
            const unsigned short* p = &As[buf][(wm * 32 + ms * 16 + fr) * 32];
            FragCast fc;
            fc.u[0] = *(const uint4*)(p + kb);
            fc.u[1] = *(const uint4*)(p + 16 + kb);
            af[ms] = fc.v;
        }
#pragma unroll
        for (int ns = 0; ns < 4; ++ns) {
            const unsigned short* p = &Bs[buf][(wn * 64 + ns * 16 + fr) * 32];
            FragCast fc;
            fc.u[0] = *(const uint4*)(p + kb);
            fc.u[1] = *(const uint4*)(p + 16 + kb);
            bfq[ns] = fc.v;
        }
#pragma unroll
        for (int ms = 0; ms < 2; ++ms)
#pragma unroll
            for (int ns = 0; ns < 4; ++ns)
                acc[ms][ns] = __builtin_amdgcn_wmma_f32_16x16x32_bf16(
                    false, af[ms], false, bfq[ns],
                    (short)0, acc[ms][ns], false, false);
        __syncthreads();
    }
#else
    uint4 ra0, ra1, rb0, rb1;
    {
        const uint4* ga = (const uint4*)aptr + half * 2;
        ra0 = ga[0]; ra1 = ga[1];
        const uint4* gb = (const uint4*)bptr + half * 2;
        rb0 = gb[0]; rb1 = gb[1];
        uint4* sa = (uint4*)&As[0][lrow * 32 + eoff]; sa[0] = ra0; sa[1] = ra1;
        uint4* sb = (uint4*)&Bs[0][lrow * 32 + eoff]; sb[0] = rb0; sb[1] = rb1;
    }
    __syncthreads();
    for (int kc = 0; kc < nk; ++kc) {
        const int buf = kc & 1;
        if (kc + 1 < nk) {
            const int k0 = (kc + 1) << 5;
            const uint4* ga = (const uint4*)(aptr + k0) + half * 2;
            ra0 = ga[0]; ra1 = ga[1];
            const uint4* gb = (const uint4*)(bptr + k0) + half * 2;
            rb0 = gb[0]; rb1 = gb[1];
            if (kc + 2 < nk) {
                __builtin_prefetch(aptr + ((kc + 2) << 5), 0, 0);
                __builtin_prefetch(bptr + ((kc + 2) << 5), 0, 0);
            }
        }
        v16bf af[2], bfq[4];
#pragma unroll
        for (int ms = 0; ms < 2; ++ms) {
            const unsigned short* p = &As[buf][(wm * 32 + ms * 16 + fr) * 32];
            FragCast fc;
            fc.u[0] = *(const uint4*)(p + kb);
            fc.u[1] = *(const uint4*)(p + 16 + kb);
            af[ms] = fc.v;
        }
#pragma unroll
        for (int ns = 0; ns < 4; ++ns) {
            const unsigned short* p = &Bs[buf][(wn * 64 + ns * 16 + fr) * 32];
            FragCast fc;
            fc.u[0] = *(const uint4*)(p + kb);
            fc.u[1] = *(const uint4*)(p + 16 + kb);
            bfq[ns] = fc.v;
        }
#pragma unroll
        for (int ms = 0; ms < 2; ++ms)
#pragma unroll
            for (int ns = 0; ns < 4; ++ns)
                acc[ms][ns] = __builtin_amdgcn_wmma_f32_16x16x32_bf16(
                    false, af[ms], false, bfq[ns],
                    (short)0, acc[ms][ns], false, false);
        if (kc + 1 < nk) {
            uint4* sa = (uint4*)&As[buf ^ 1][lrow * 32 + eoff]; sa[0] = ra0; sa[1] = ra1;
            uint4* sb = (uint4*)&Bs[buf ^ 1][lrow * 32 + eoff]; sb[0] = rb0; sb[1] = rb1;
        }
        __syncthreads();
    }
#endif

    // argmin epilogue (deterministic u64 atomicMin, low-index tiebreak)
#pragma unroll
    for (int ms = 0; ms < 2; ++ms) {
        const int rbase = m0 + wm * 32 + ms * 16 + (lane >> 4) * 8;
#pragma unroll
        for (int i = 0; i < 8; ++i) {
            unsigned long long best = ~0ull;
#pragma unroll
            for (int ns = 0; ns < 4; ++ns) {
                const int col = n0 + wn * 64 + ns * 16 + fr;
                float dv = enorm[col] - 2.f * acc[ms][ns][i];
                unsigned u = __float_as_uint(dv);
                u = (u & 0x80000000u) ? ~u : (u | 0x80000000u);
                unsigned long long key =
                    ((unsigned long long)u << 32) | (unsigned long long)(unsigned)col;
                best = key < best ? key : best;
            }
            for (int off = 8; off > 0; off >>= 1) {
                unsigned long long o = shfl_xor_u64(best, off);
                best = o < best ? o : best;
            }
            if ((lane & 15) == 0)
                atomicMin(&keys[rbase + i], best);
        }
    }
}

// ---------------------------------------------------------------------------
__global__ void init_keys_kernel(unsigned long long* keys) {
    int i = blockIdx.x * 256 + threadIdx.x;
    if (i < VB) keys[i] = ~0ull;
}

__global__ void enorm_kernel(const float* __restrict__ emb, float* __restrict__ enorm) {
    int k = blockIdx.x * 256 + threadIdx.x;
    if (k < VK) {
        const float* r = emb + (size_t)k * VD;
        float s = 0.f;
        for (int d = 0; d < VD; d += 4) {
            float4 f = *(const float4*)(r + d);
            s += f.x * f.x + f.y * f.y + f.z * f.z + f.w * f.w;
        }
        enorm[k] = s;
    }
}

__global__ __launch_bounds__(256)
void vq_loss_kernel(const float* __restrict__ ze, const float* __restrict__ emb,
                    const unsigned long long* __restrict__ keys,
                    float* __restrict__ vq_partials)
{
    __shared__ float red[256];
    const int b = blockIdx.x * 256 + threadIdx.x;
    const float* zr = ze + (size_t)b * VD;
    const unsigned idx = (unsigned)(keys[b] & 0xffffffffull);
    const float* er = emb + (size_t)idx * VD;
    float s = 0.f;
    for (int d = 0; d < VD; d += 4) {
        float4 a = *(const float4*)(zr + d);
        float4 e = *(const float4*)(er + d);
        float dx = a.x - e.x, dy = a.y - e.y, dz = a.z - e.z, dw = a.w - e.w;
        s += dx * dx + dy * dy + dz * dz + dw * dw;
    }
    red[threadIdx.x] = s;
    __syncthreads();
    for (int st = 128; st > 0; st >>= 1) {
        if (threadIdx.x < st) red[threadIdx.x] += red[threadIdx.x + st];
        __syncthreads();
    }
    if (threadIdx.x == 0) vq_partials[blockIdx.x] = red[0];
}

__global__ void finalize_kernel(const float* __restrict__ bce_p, int nbce,
                                const float* __restrict__ vq_p, int nvq,
                                float* __restrict__ out3)
{
    if (threadIdx.x == 0 && blockIdx.x == 0) {
        double sb = 0.0;
        for (int i = 0; i < nbce; ++i) sb += (double)bce_p[i];
        double sv = 0.0;
        for (int i = 0; i < nvq; ++i) sv += (double)vq_p[i];
        out3[0] = (float)(-sb / ((double)VB * (double)VDIN));  // reconst_loss
        float vl = (float)(sv / (double)VB);
        out3[1] = vl;                                          // embed_loss
        out3[2] = vl;                                          // commit_loss
    }
}

// ---------------------------------------------------------------------------
extern "C" void kernel_launch(void* const* d_in, const int* in_sizes, int n_in,
                              void* d_out, int out_size, void* d_ws, size_t ws_size,
                              hipStream_t stream) {
    (void)in_sizes; (void)n_in; (void)out_size; (void)ws_size;
    const float* x   = (const float*)d_in[0];
    const float* W1  = (const float*)d_in[1];
    const float* b1  = (const float*)d_in[2];
    const float* W2  = (const float*)d_in[3];
    const float* b2  = (const float*)d_in[4];
    const float* W3  = (const float*)d_in[5];
    const float* b3  = (const float*)d_in[6];
    const float* W4  = (const float*)d_in[7];
    const float* b4  = (const float*)d_in[8];
    const float* emb = (const float*)d_in[9];
    float* out = (float*)d_out;

    char* ws = (char*)d_ws;
    size_t off = 0;
    auto take = [&](size_t bytes) -> void* {
        void* p = (void*)(ws + off);
        off += (bytes + 255) & ~(size_t)255;
        return p;
    };

    const int gn4  = NP_O / 128;                     // 7
    const int nbce = gn4 * (VB / 128);               // 896
    const int nvq  = VB / 256;                       // 64

    unsigned short* xb   = (unsigned short*)take((size_t)VB * KP1 * 2);
    unsigned short* embb = (unsigned short*)take((size_t)VK * VD * 2);
    unsigned short* h1b  = (unsigned short*)take((size_t)VB * KP2 * 2);
    unsigned short* zeb  = (unsigned short*)take((size_t)VB * VD * 2);
    unsigned short* h3b  = (unsigned short*)take((size_t)VB * KP2 * 2);
    float*          zef  = (float*)take((size_t)VB * VD * 4);
    unsigned short* W1t  = (unsigned short*)take((size_t)NP_H * KP1 * 2);
    unsigned short* W2t  = (unsigned short*)take((size_t)NP_H * KP2 * 2);
    unsigned short* W3t  = (unsigned short*)take((size_t)NP_H * VD  * 2);
    unsigned short* W4t  = (unsigned short*)take((size_t)NP_O * KP2 * 2);
    unsigned long long* keys = (unsigned long long*)take((size_t)VB * 8);
    float* enorm = (float*)take((size_t)VK * 4);
    float* bce_p = (float*)take((size_t)nbce * 4);
    float* vq_p  = (float*)take((size_t)nvq * 4);

    // ---- one-time conversions for this call ----
    {
        int t = VB * (KP1 / 4);
        convert_pad_kernel<<<(t + 255) / 256, 256, 0, stream>>>(x, xb, VDIN, KP1, t);
    }
    {
        int t = VK * (VD / 4);
        convert_pad_kernel<<<(t + 255) / 256, 256, 0, stream>>>(emb, embb, VD, VD, t);
    }
    {
        int t = NP_H * (KP1 / 2);
        transpose_convert_kernel<<<(t + 255) / 256, 256, 0, stream>>>(W1, W1t, VDIN, VH, KP1, t);
    }
    {
        int t = NP_H * (KP2 / 2);
        transpose_convert_kernel<<<(t + 255) / 256, 256, 0, stream>>>(W2, W2t, VH, VD, KP2, t);
    }
    {
        int t = NP_H * (VD / 2);
        transpose_convert_kernel<<<(t + 255) / 256, 256, 0, stream>>>(W3, W3t, VD, VH, VD, t);
    }
    {
        int t = NP_O * (KP2 / 2);
        transpose_convert_kernel<<<(t + 255) / 256, 256, 0, stream>>>(W4, W4t, VH, VDIN, KP2, t);
    }
    init_keys_kernel<<<VB / 256, 256, 0, stream>>>(keys);
    enorm_kernel<<<VK / 256, 256, 0, stream>>>(emb, enorm);

    // h1 = relu(x @ W1 + b1)
    gemm_bf16_kernel<ACT_RELU, false, false, true>
        <<<dim3(NP_H / 128, VB / 128), 256, 0, stream>>>(
            xb, KP1, nullptr, W1t, KP1, b1,
            nullptr, h1b, KP2, VH, KP1 / 32, nullptr, nullptr);
    // z_e = h1 @ W2 + b2
    gemm_bf16_kernel<ACT_NONE, false, true, true>
        <<<dim3(NP_H / 128, VB / 128), 256, 0, stream>>>(
            h1b, KP2, nullptr, W2t, KP2, b2,
            zef, zeb, VD, VD, KP2 / 32, nullptr, nullptr);
    // argmin_k ||z_e - emb_k||^2
    vq_dist_argmin_kernel<<<dim3(VK / 128, VB / 128), 256, 0, stream>>>(
        zeb, embb, enorm, keys);
    // embed/commit loss partials
    vq_loss_kernel<<<nvq, 256, 0, stream>>>(zef, emb, keys, vq_p);
    // h3 = relu(emb[idx] @ W3 + b3)
    gemm_bf16_kernel<ACT_RELU, true, false, true>
        <<<dim3(NP_H / 128, VB / 128), 256, 0, stream>>>(
            embb, VD, keys, W3t, VD, b3,
            nullptr, h3b, KP2, VH, VD / 32, nullptr, nullptr);
    // x_reconst = sigmoid(h3 @ W4 + b4), fused BCE partials
    gemm_bf16_kernel<ACT_SIGBCE, false, true, false>
        <<<dim3(gn4, VB / 128), 256, 0, stream>>>(
            h3b, KP2, nullptr, W4t, KP2, b4,
            out, nullptr, 0, VDIN, KP2 / 32, x, bce_p);
    // scalars
    finalize_kernel<<<1, 32, 0, stream>>>(bce_p, nbce, vq_p, nvq,
                                          out + (size_t)VB * VDIN);
}
